// Transformer_8976481649165
// MI455X (gfx1250) — matrix operand order repaired and checked by reading, not compiled, not run on hardware
//
#include <hip/hip_runtime.h>

// ---------------- problem constants (from reference) ----------------
#define DI  1024      // model dim
#define HH  16        // heads
#define HDI 64        // head dim
#define LLY 6         // layers
#define FFD 4096      // ffn dim
#define VSZ 32000     // vocab
#define BB  4
#define SSQ 512       // src len
#define TSQ 512       // trg len
#define MROWS (BB*SSQ)   // 2048 rows for every GEMM

typedef __attribute__((ext_vector_type(16))) _Float16 v16h;
typedef __attribute__((ext_vector_type(8)))  float    v8f;

// ---------------- WMMA fragment helpers (wave32, 16x16x32 f16) ------
// A (16x32, MxK): lane L -> row M = L%16, khalf = L/16.
//   element e -> K = k0 + khalf*16 + e        (contiguous 16 floats)
static __device__ __forceinline__ v16h ld_a(const float* __restrict__ p) {
  v16h a;
#pragma unroll
  for (int e = 0; e < 8; ++e) a[e] = (_Float16)p[e];
#pragma unroll
  for (int e = 0; e < 8; ++e) a[e + 8] = (_Float16)p[e + 8];
  return a;
}
// B (32x16, KxN): lane L -> col N = L%16, khalf = L/16; element e -> K = k0 + khalf*16 + e.
// Row-major W[K][N]: stride-N gather.
static __device__ __forceinline__ v16h ld_b_strided(const float* __restrict__ p, int ld) {
  v16h b;
#pragma unroll
  for (int e = 0; e < 16; ++e) b[e] = (_Float16)p[(size_t)e * ld];
  return b;
}
// B fragment when the "B matrix" is stored transposed (K contiguous), e.g. K^T in attention.
static __device__ __forceinline__ v16h ld_b_contig(const float* __restrict__ p) {
  v16h b;
#pragma unroll
  for (int e = 0; e < 16; ++e) b[e] = (_Float16)p[e];
  return b;
}
static __device__ __forceinline__ v8f wmma16(v16h a, v16h b, v8f c) {
  return __builtin_amdgcn_wmma_f32_16x16x32_f16(false, a, false, b, (short)0, c, false, false);
}

// ---------------- generic GEMM: C = act(A[MxK] @ W[KxN] + bias) -----
// block = 128 threads = 4 waves; each wave computes a 16x64 strip (4 accumulators)
// so the A fragment is reused across 4 WMMAs per K-step.
// grid = (N/256, M/16)
__global__ void __launch_bounds__(128)
gemm_bias_act(const float* __restrict__ A, const float* __restrict__ W,
              const float* __restrict__ bias, float* __restrict__ C,
              int M, int N, int K, int relu)
{
  const int lane  = threadIdx.x & 31;
  const int khalf = lane >> 4;
  const int l16   = lane & 15;
  const int n0 = (blockIdx.x * 4 + (threadIdx.x >> 5)) * 64;
  const int m0 = blockIdx.y * 16;
  const float* arow  = A + (size_t)(m0 + l16) * K;
  const float* wbase = W + n0 + l16;
  v8f c0 = {}, c1 = {}, c2 = {}, c3 = {};
  for (int k0 = 0; k0 < K; k0 += 32) {
    __builtin_prefetch(arow + k0 + 256, 0, 0);   // global_prefetch_b8 (speculative)
    v16h a = ld_a(arow + k0 + khalf * 16);
    const float* wk = wbase + (size_t)(k0 + khalf * 16) * N;
    v16h b0 = ld_b_strided(wk,      N);
    v16h b1 = ld_b_strided(wk + 16, N);
    v16h b2 = ld_b_strided(wk + 32, N);
    v16h b3 = ld_b_strided(wk + 48, N);
    c0 = wmma16(a, b0, c0);
    c1 = wmma16(a, b1, c1);
    c2 = wmma16(a, b2, c2);
    c3 = wmma16(a, b3, c3);
  }
  // C/D layout: lane L holds M = m0 + j + 8*(L/16), N = tile_n + L%16
  float* crow = C + (size_t)(m0 + 8 * khalf) * N + n0 + l16;
#pragma unroll
  for (int t = 0; t < 4; ++t) {
    const v8f cc = (t == 0) ? c0 : (t == 1) ? c1 : (t == 2) ? c2 : c3;
    const float bv = bias ? bias[n0 + t * 16 + l16] : 0.0f;
#pragma unroll
    for (int j = 0; j < 8; ++j) {
      float v = cc[j] + bv;
      if (relu) v = fmaxf(v, 0.0f);
      crow[(size_t)j * N + t * 16] = v;
    }
  }
}

// ---------------- attention: scores = (Q K^T) / 8 -------------------
// Q: [B, Lq, D], K: [B, Lk, D] (head h = cols h*64..h*64+63); scores: [B,H,Lq,Lk]
// block = 256 = 8 waves; each wave emits a 16x64 strip of scores (Lk = 512 total).
// grid = (1, Lq/16, B*H)
__global__ void __launch_bounds__(256)
attn_scores(const float* __restrict__ Q, const float* __restrict__ Kc,
            float* __restrict__ sc, int Lq, int Lk)
{
  const int lane  = threadIdx.x & 31;
  const int khalf = lane >> 4;
  const int l16   = lane & 15;
  const int bh = blockIdx.z;
  const int b  = bh / HH;
  const int h  = bh % HH;
  const int n0 = (threadIdx.x >> 5) * 64;     // 8 waves cover Lk = 512
  const int m0 = blockIdx.y * 16;
  const float* Qb = Q + ((size_t)b * Lq + m0 + l16) * DI + h * HDI;
  const float* K0 = Kc + ((size_t)b * Lk + n0 +  0 + l16) * DI + h * HDI;
  const float* K1 = Kc + ((size_t)b * Lk + n0 + 16 + l16) * DI + h * HDI;
  const float* K2 = Kc + ((size_t)b * Lk + n0 + 32 + l16) * DI + h * HDI;
  const float* K3 = Kc + ((size_t)b * Lk + n0 + 48 + l16) * DI + h * HDI;
  v8f c0 = {}, c1 = {}, c2 = {}, c3 = {};
#pragma unroll
  for (int k0 = 0; k0 < HDI; k0 += 32) {
    v16h a = ld_a(Qb + k0 + khalf * 16);
    const int ko = k0 + khalf * 16;
    c0 = wmma16(a, ld_b_contig(K0 + ko), c0);
    c1 = wmma16(a, ld_b_contig(K1 + ko), c1);
    c2 = wmma16(a, ld_b_contig(K2 + ko), c2);
    c3 = wmma16(a, ld_b_contig(K3 + ko), c3);
  }
  float* srow = sc + ((size_t)bh * Lq + m0 + 8 * khalf) * Lk + n0 + l16;
#pragma unroll
  for (int t = 0; t < 4; ++t) {
    const v8f cc = (t == 0) ? c0 : (t == 1) ? c1 : (t == 2) ? c2 : c3;
#pragma unroll
    for (int j = 0; j < 8; ++j)
      srow[(size_t)j * Lk + t * 16] = cc[j] * 0.125f;
  }
}

// ---------------- masked softmax over rows of length Lk -------------
// one wave per row; causal=1: valid = (trg[b,q]!=0 && k<=q); else valid = (src[b,k]!=0)
// grid = (B*H*Lq/8), block 256
__global__ void __launch_bounds__(256)
softmax_mask(float* __restrict__ sc, const int* __restrict__ toks,
             int Lq, int Lk, int causal)
{
  const int lane = threadIdx.x & 31;
  const int row  = blockIdx.x * 8 + (threadIdx.x >> 5);   // index into B*H*Lq
  const int q  = row % Lq;
  const int bh = row / Lq;
  const int b  = bh / HH;
  float* p = sc + (size_t)row * Lk;
  const int PER = Lk / 32;        // 16
  float vals[16];
  float mx = -3.0e38f;
  const int qvalid = causal ? (toks[b * Lq + q] != 0) : 1;
#pragma unroll
  for (int i = 0; i < PER; ++i) {
    int k = i * 32 + lane;
    int valid = causal ? (qvalid && (k <= q)) : (toks[b * Lk + k] != 0);
    float v = valid ? p[k] : -1.0e9f;
    vals[i] = v;
    mx = fmaxf(mx, v);
  }
#pragma unroll
  for (int off = 16; off; off >>= 1) mx = fmaxf(mx, __shfl_xor(mx, off, 32));
  float sum = 0.0f;
#pragma unroll
  for (int i = 0; i < PER; ++i) { vals[i] = __expf(vals[i] - mx); sum += vals[i]; }
#pragma unroll
  for (int off = 16; off; off >>= 1) sum += __shfl_xor(sum, off, 32);
  const float inv = 1.0f / sum;
#pragma unroll
  for (int i = 0; i < PER; ++i) p[i * 32 + lane] = vals[i] * inv;
}

// ---------------- attention: Out = P @ V ----------------------------
// P: [B,H,Lq,Lk], V: [B,Lk,D]; Out: [B,Lq,D] (heads concatenated)
// block = 256 = 8 waves; wave w handles query tile (blockIdx.y*8 + w) and the
// full 64-wide head dim (4 accumulators; P fragment reused across all 4).
// grid = (1, Lq/128, B*H)
__global__ void __launch_bounds__(256)
attn_pv(const float* __restrict__ P, const float* __restrict__ Vc,
        float* __restrict__ Out, int Lq, int Lk)
{
  const int lane  = threadIdx.x & 31;
  const int khalf = lane >> 4;
  const int l16   = lane & 15;
  const int bh = blockIdx.z;
  const int b  = bh / HH;
  const int h  = bh % HH;
  const int m0 = (blockIdx.y * 8 + (threadIdx.x >> 5)) * 16;
  const int n0 = h * HDI;
  const float* Pb = P + ((size_t)bh * Lq + m0 + l16) * Lk;
  const float* Vb = Vc + (size_t)b * Lk * DI + n0 + l16;
  v8f c0 = {}, c1 = {}, c2 = {}, c3 = {};
  for (int k0 = 0; k0 < Lk; k0 += 32) {
    __builtin_prefetch(Pb + k0 + 256, 0, 0);
    v16h a = ld_a(Pb + k0 + khalf * 16);
    const float* vk = Vb + (size_t)(k0 + khalf * 16) * DI;
    c0 = wmma16(a, ld_b_strided(vk,      DI), c0);
    c1 = wmma16(a, ld_b_strided(vk + 16, DI), c1);
    c2 = wmma16(a, ld_b_strided(vk + 32, DI), c2);
    c3 = wmma16(a, ld_b_strided(vk + 48, DI), c3);
  }
  float* orow = Out + ((size_t)b * Lq + m0 + 8 * khalf) * DI + n0 + l16;
#pragma unroll
  for (int t = 0; t < 4; ++t) {
    const v8f cc = (t == 0) ? c0 : (t == 1) ? c1 : (t == 2) ? c2 : c3;
#pragma unroll
    for (int j = 0; j < 8; ++j)
      orow[(size_t)j * DI + t * 16] = cc[j];
  }
}

// ---------------- fused residual + LayerNorm ------------------------
// out[row] = LN(resid[row] + delta[row]) * g + beta ; one wave per row, D=1024
// grid = (rows/8), block 256. out may alias resid.
__global__ void __launch_bounds__(256)
add_layernorm(const float* __restrict__ resid, const float* __restrict__ delta,
              const float* __restrict__ g, const float* __restrict__ beta,
              float* __restrict__ out)
{
  const int lane = threadIdx.x & 31;
  const int row  = blockIdx.x * 8 + (threadIdx.x >> 5);
  const float* r  = resid + (size_t)row * DI;
  const float* dl = delta + (size_t)row * DI;
  float vals[32];
  float s = 0.0f;
#pragma unroll
  for (int i = 0; i < 32; ++i) {
    int d = i * 32 + lane;
    float v = r[d] + dl[d];
    vals[i] = v; s += v;
  }
#pragma unroll
  for (int off = 16; off; off >>= 1) s += __shfl_xor(s, off, 32);
  const float mu = s * (1.0f / DI);
  float vs = 0.0f;
#pragma unroll
  for (int i = 0; i < 32; ++i) { float t = vals[i] - mu; vs += t * t; }
#pragma unroll
  for (int off = 16; off; off >>= 1) vs += __shfl_xor(vs, off, 32);
  const float rstd = rsqrtf(vs * (1.0f / DI) + 1e-5f);
  float* o = out + (size_t)row * DI;
#pragma unroll
  for (int i = 0; i < 32; ++i) {
    int d = i * 32 + lane;
    o[d] = (vals[i] - mu) * rstd * g[d] + beta[d];
  }
}

// ---------------- token embedding + sinusoidal PE -------------------
// out[b,s,d] = emb[tok[b,s], d] + PE(s, d); grid covers B*seq*D elements
__global__ void __launch_bounds__(256)
embed_pe(const int* __restrict__ tok, const float* __restrict__ emb,
         float* __restrict__ out, int seq)
{
  const int idx = blockIdx.x * blockDim.x + threadIdx.x;
  const int d = idx % DI;
  const int sflat = idx / DI;     // b*seq + s
  const int s = sflat % seq;
  const int t = tok[sflat];
  const float i2  = (float)(d & ~1);
  const float ang = (float)s * __expf(-(i2 / (float)DI) * 9.2103403719761836f); // ln(10000)
  const float pe  = (d & 1) ? __cosf(ang) : __sinf(ang);
  out[idx] = emb[(size_t)t * DI + d] + pe;
}

// ---------------- host orchestration --------------------------------
static inline void gemm(const float* A, const float* W, const float* bias, float* C,
                        int M, int N, int K, int relu, hipStream_t s)
{
  dim3 grid(N / 256, M / 16, 1);
  gemm_bias_act<<<grid, 128, 0, s>>>(A, W, bias, C, M, N, K, relu);
}

extern "C" void kernel_launch(void* const* d_in, const int* in_sizes, int n_in,
                              void* d_out, int out_size, void* d_ws, size_t ws_size,
                              hipStream_t stream)
{
  (void)in_sizes; (void)n_in; (void)out_size; (void)ws_size;

  // ---- decode inputs in setup_inputs() insertion order ----
  const int*   src     = (const int*)d_in[0];
  const int*   trg     = (const int*)d_in[1];
  const float* emb_enc = (const float*)d_in[2];
  const float* emb_dec = (const float*)d_in[3];
  int idx = 4;
  // enc layer: self(Wq,Wk,Wv,Wo)=0..3, ff(W1,b1,W2,b2)=4..7, n1(g,b)=8..9, n2(g,b)=10..11
  const float* encp[LLY][12];
  for (int l = 0; l < LLY; ++l)
    for (int j = 0; j < 12; ++j) encp[l][j] = (const float*)d_in[idx++];
  // dec layer: self=0..3, cross=4..7, ff=8..11, n1=12..13, n2=14..15, n3=16..17
  const float* decp[LLY][18];
  for (int l = 0; l < LLY; ++l)
    for (int j = 0; j < 18; ++j) decp[l][j] = (const float*)d_in[idx++];
  const float* fcW = (const float*)d_in[idx++];
  const float* fcb = (const float*)d_in[idx++];

  // ---- workspace layout (floats) ----
  float* w = (float*)d_ws;
  const size_t ND  = (size_t)MROWS * DI;          // 2M floats
  const size_t NF  = (size_t)MROWS * FFD;         // 8M floats
  const size_t NSC = (size_t)BB * HH * SSQ * SSQ; // 16.8M floats
  float* xe    = w;              w += ND;   // encoder state -> enc_out
  float* yd    = w;              w += ND;   // decoder state
  float* qb    = w;              w += ND;
  float* kb    = w;              w += ND;
  float* vb    = w;              w += ND;
  float* attnb = w;              w += ND;
  float* tmp   = w;              w += ND;
  float* mid   = w;              w += NF;
  float* sc    = w;              w += NSC;

  const int M = MROWS;
  const dim3 sgrid(1, SSQ / 16, BB * HH);    // attn_scores grid (8 waves cover Lk=512)
  const dim3 pgrid(1, SSQ / 128, BB * HH);   // attn_pv grid (8 waves cover 8 q-tiles)
  const int  softb = BB * HH * SSQ / 8;      // softmax blocks
  const int  lnb   = M / 8;                  // layernorm blocks
  const int  embb  = (int)(ND / 256);        // embed blocks

  // ================= encoder =================
  embed_pe<<<embb, 256, 0, stream>>>(src, emb_enc, xe, SSQ);
  for (int l = 0; l < LLY; ++l) {
    const float* const* p = encp[l];
    gemm(xe, p[0], nullptr, qb, M, DI, DI, 0, stream);   // Wq
    gemm(xe, p[1], nullptr, kb, M, DI, DI, 0, stream);   // Wk
    gemm(xe, p[2], nullptr, vb, M, DI, DI, 0, stream);   // Wv
    attn_scores<<<sgrid, 256, 0, stream>>>(qb, kb, sc, SSQ, SSQ);
    softmax_mask<<<softb, 256, 0, stream>>>(sc, src, SSQ, SSQ, 0);
    attn_pv<<<pgrid, 256, 0, stream>>>(sc, vb, attnb, SSQ, SSQ);
    gemm(attnb, p[3], nullptr, tmp, M, DI, DI, 0, stream);  // Wo
    add_layernorm<<<lnb, 256, 0, stream>>>(xe, tmp, p[8], p[9], xe);   // n1
    gemm(xe, p[4], p[5], mid, M, FFD, DI, 1, stream);    // W1 + b1, relu
    gemm(mid, p[6], p[7], tmp, M, DI, FFD, 0, stream);   // W2 + b2
    add_layernorm<<<lnb, 256, 0, stream>>>(xe, tmp, p[10], p[11], xe); // n2
  }
  // xe now holds enc_out

  // ================= decoder =================
  embed_pe<<<embb, 256, 0, stream>>>(trg, emb_dec, yd, TSQ);
  for (int l = 0; l < LLY; ++l) {
    const float* const* p = decp[l];
    // --- masked self-attention ---
    gemm(yd, p[0], nullptr, qb, M, DI, DI, 0, stream);
    gemm(yd, p[1], nullptr, kb, M, DI, DI, 0, stream);
    gemm(yd, p[2], nullptr, vb, M, DI, DI, 0, stream);
    attn_scores<<<sgrid, 256, 0, stream>>>(qb, kb, sc, TSQ, TSQ);
    softmax_mask<<<softb, 256, 0, stream>>>(sc, trg, TSQ, TSQ, 1);
    attn_pv<<<pgrid, 256, 0, stream>>>(sc, vb, attnb, TSQ, TSQ);
    gemm(attnb, p[3], nullptr, tmp, M, DI, DI, 0, stream);
    add_layernorm<<<lnb, 256, 0, stream>>>(yd, tmp, p[12], p[13], yd); // n1
    // --- cross-attention (K,V from enc_out) ---
    gemm(yd, p[4], nullptr, qb, M, DI, DI, 0, stream);
    gemm(xe, p[5], nullptr, kb, M, DI, DI, 0, stream);
    gemm(xe, p[6], nullptr, vb, M, DI, DI, 0, stream);
    attn_scores<<<sgrid, 256, 0, stream>>>(qb, kb, sc, TSQ, SSQ);
    softmax_mask<<<softb, 256, 0, stream>>>(sc, src, TSQ, SSQ, 0);
    attn_pv<<<pgrid, 256, 0, stream>>>(sc, vb, attnb, TSQ, SSQ);
    gemm(attnb, p[7], nullptr, tmp, M, DI, DI, 0, stream);
    add_layernorm<<<lnb, 256, 0, stream>>>(yd, tmp, p[14], p[15], yd); // n2
    // --- FFN ---
    gemm(yd, p[8], p[9], mid, M, FFD, DI, 1, stream);
    gemm(mid, p[10], p[11], tmp, M, DI, FFD, 0, stream);
    add_layernorm<<<lnb, 256, 0, stream>>>(yd, tmp, p[16], p[17], yd); // n3
  }

  // ================= final projection to vocab =================
  gemm(yd, fcW, fcb, (float*)d_out, M, VSZ, DI, 0, stream);
}
